// CapsuleNetwork_29137058136441
// MI455X (gfx1250) — matrix operand order repaired and checked by reading, not compiled, not run on hardware
//
#include <hip/hip_runtime.h>

// ---------------------------------------------------------------------------
// CapsNet forward for MI455X (gfx1250, wave32).
// conv2 + decoder GEMMs: v_wmma_f32_16x16x32_bf16, double-buffered LDS A tile
// filled with global_load_async_to_lds_b128 (ASYNCcnt), B fragments loaded
// directly from global in pre-swizzled WMMA lane layout (no LDS, no transpose).
// ---------------------------------------------------------------------------

typedef __attribute__((ext_vector_type(16))) __bf16 v16bf;
typedef __attribute__((ext_vector_type(8)))  float  v8f;

union FragBF { v16bf v; unsigned int u[8]; };

__device__ __forceinline__ unsigned short f32_to_bf16(float f) {
  unsigned int u = __float_as_uint(f);
  u += 0x7FFFu + ((u >> 16) & 1u);      // round-to-nearest-even
  return (unsigned short)(u >> 16);
}

// gfx1250 async global->LDS copy, 16B per lane. VDST = LDS byte address
// (low 32 bits of the generic pointer), VADDR = 64-bit global address.
__device__ __forceinline__ void async_load_b128(unsigned lds_addr,
                                                const void* gaddr) {
  asm volatile("global_load_async_to_lds_b128 %0, %1, off"
               :: "v"(lds_addr), "v"(gaddr) : "memory");
}
__device__ __forceinline__ void wait_async0() {
  asm volatile("s_wait_asynccnt 0" ::: "memory");
}

// Load a 16x16x32 bf16 A fragment from an LDS tile laid out [row][k] (32 k).
__device__ __forceinline__ FragBF load_fragA(const unsigned short* tile,
                                             int row, int lane) {
  FragBF fa;
  const int kb = (lane < 16) ? 0 : 16;  // lane<16: K{0..15}, else {16..31}
  const uint4 q0 = *reinterpret_cast<const uint4*>(&tile[row * 32 + kb]);
  const uint4 q1 = *reinterpret_cast<const uint4*>(&tile[row * 32 + kb + 8]);
  fa.u[0] = q0.x; fa.u[1] = q0.y; fa.u[2] = q0.z; fa.u[3] = q0.w;
  fa.u[4] = q1.x; fa.u[5] = q1.y; fa.u[6] = q1.z; fa.u[7] = q1.w;
  return fa;
}
// Load a B fragment (32 bytes/lane) directly from pre-swizzled global memory.
__device__ __forceinline__ FragBF load_fragB(const unsigned short* wb) {
  FragBF fb;
  const uint4 q0 = *reinterpret_cast<const uint4*>(wb);
  const uint4 q1 = *reinterpret_cast<const uint4*>(wb + 8);
  fb.u[0] = q0.x; fb.u[1] = q0.y; fb.u[2] = q0.z; fb.u[3] = q0.w;
  fb.u[4] = q1.x; fb.u[5] = q1.y; fb.u[6] = q1.z; fb.u[7] = q1.w;
  return fb;
}

// ---------------------------------------------------------------------------
// conv1: [256,1,28,28] -> ReLU -> bf16 xT[b, pos(20x20), oc(256)]  (ic-last)
// ---------------------------------------------------------------------------
__global__ void __launch_bounds__(256) conv1_kernel(
    const float* __restrict__ in, const float* __restrict__ w,
    const float* __restrict__ bias, unsigned short* __restrict__ xT) {
  __shared__ float img[784];
  const int b  = blockIdx.x;
  const int oc = threadIdx.x;
  for (int i = threadIdx.x; i < 784; i += 256) img[i] = in[b * 784 + i];
  float wreg[81];
#pragma unroll
  for (int i = 0; i < 81; ++i) wreg[i] = w[oc * 81 + i];
  const float bs = bias[oc];
  __syncthreads();
  for (int pos = 0; pos < 400; ++pos) {
    const int oy = pos / 20, ox = pos % 20;
    float acc = bs;
#pragma unroll
    for (int ky = 0; ky < 9; ++ky)
#pragma unroll
      for (int kx = 0; kx < 9; ++kx)
        acc = fmaf(img[(oy + ky) * 28 + ox + kx], wreg[ky * 9 + kx], acc);
    acc = fmaxf(acc, 0.0f);
    xT[(b * 400 + pos) * 256 + oc] = f32_to_bf16(acc);
  }
}

// ---------------------------------------------------------------------------
// Pack conv2 weights into WMMA B-fragment layout:
//   wf[(kk*16 + nt)*512 + lane*16 + h], kk = kp*8+kc (0..647), nt = oc/16.
//   lane l covers oc = nt*16 + (l&15), k = kc*32 + (l<16?0:16) + h.
// ---------------------------------------------------------------------------
__global__ void pack_w2_kernel(const float* __restrict__ w2,
                               unsigned short* __restrict__ wf) {
  const int idx = blockIdx.x * 256 + threadIdx.x;   // exactly 81*65536
  const int h  = idx & 15;
  const int l  = (idx >> 4) & 31;
  const int nt = (idx >> 9) & 15;
  const int kc = (idx >> 13) & 7;
  const int kp = idx >> 16;
  const int oc = nt * 16 + (l & 15);
  const int ic = kc * 32 + ((l < 16) ? 0 : 16) + h;
  wf[idx] = f32_to_bf16(w2[(oc * 256 + ic) * 81 + kp]);
}

// Pack decoder weights src[N][K] into the same B-fragment layout with
// ldn (multiple of 64) columns, zero-padded for n >= N.
__global__ void pack_dec_kernel(const float* __restrict__ src,
                                unsigned short* __restrict__ dst,
                                int N, int K, int ldn) {
  const int idx = blockIdx.x * 256 + threadIdx.x;
  if (idx >= K * ldn) return;
  const int h    = idx & 15;
  const int l    = (idx >> 4) & 31;
  const int rest = idx >> 9;
  const int nt_total = ldn >> 4;
  const int nt = rest % nt_total;
  const int kc = rest / nt_total;
  const int n  = nt * 16 + (l & 15);
  const int k  = kc * 32 + ((l < 16) ? 0 : 16) + h;
  dst[idx] = f32_to_bf16(n < N ? src[n * K + k] : 0.0f);
}

// ---------------------------------------------------------------------------
// conv2 implicit GEMM: M=9216 (b,oy,ox), N=256 (oc), K=648*32.
// Block tile 64x64; 8 waves (4M x 2N); each wave: 2 accumulators.
// A tile double-buffered in LDS via async loads; B fragments from global.
// ---------------------------------------------------------------------------
__global__ void __launch_bounds__(256) conv2_wmma_kernel(
    const unsigned short* __restrict__ xT, const unsigned short* __restrict__ wf,
    const float* __restrict__ bias, float* __restrict__ y) {
  __shared__ __align__(16) unsigned short As[2][64 * 32];
  __shared__ int rowbase[64];
  const int tid  = threadIdx.x;
  const int m0   = blockIdx.x * 64;
  const int nblk = blockIdx.y;                 // 0..3 -> oc block of 64
  if (tid < 64) {
    const int m = m0 + tid;
    const int b = m / 36, s = m % 36;
    rowbase[tid] = (b * 400 + (s / 6) * 40 + (s % 6) * 2) * 256;
  }
  const int lane = tid & 31;
  const int wave = tid >> 5;
  const int wm = wave & 3, wn = wave >> 2;     // 4 x 2 wave grid
  const int ar = tid >> 2, ac = tid & 3;       // A stage: 64 rows x 4 x 16B
  v8f acc0 = {}, acc1 = {};
  __syncthreads();

  const unsigned ldsA0 = (unsigned)(size_t)&As[0][ar * 32 + ac * 8];
  const unsigned ldsA1 = (unsigned)(size_t)&As[1][ar * 32 + ac * 8];
  const unsigned short* gA = xT + rowbase[ar] + ac * 8;
  const int arow = wm * 16 + (lane & 15);

  const int KTOT = 648;                        // 81 kernel pos x 8 ic-chunks
  async_load_b128(ldsA0, gA);                  // k-step 0
  wait_async0();
  __syncthreads();

  for (int kk = 0; kk < KTOT; ++kk) {
    const unsigned short* cur = As[kk & 1];
    if (kk + 1 < KTOT) {                       // prefetch next A tile (async)
      async_load_b128((kk & 1) ? ldsA0 : ldsA1, gA + (kk + 1) * 32);
      __builtin_prefetch(wf + ((size_t)(kk + 1) * 16 + nblk * 4 + wn * 2) * 512
                            + lane * 16, 0, 1);
    }
    const FragBF fa = load_fragA(cur, arow, lane);
    const unsigned short* wb =
        wf + ((size_t)kk * 16 + nblk * 4 + wn * 2) * 512 + lane * 16;
    const FragBF fb0 = load_fragB(wb);
    const FragBF fb1 = load_fragB(wb + 512);
    acc0 = __builtin_amdgcn_wmma_f32_16x16x32_bf16(
        false, fa.v, false, fb0.v, (short)0, acc0, false, false);
    acc1 = __builtin_amdgcn_wmma_f32_16x16x32_bf16(
        false, fa.v, false, fb1.v, (short)0, acc1, false, false);
    wait_async0();
    __syncthreads();
  }

  const int nt0 = nblk * 4 + wn * 2;
#pragma unroll
  for (int s = 0; s < 2; ++s) {
    const int n = (nt0 + s) * 16 + (lane & 15);
    const float bs = bias[n];
    const v8f a = s ? acc1 : acc0;
#pragma unroll
    for (int r = 0; r < 8; ++r) {
      const int m = m0 + wm * 16 + r + ((lane < 16) ? 0 : 8);
      const int b = m / 36, sp = m % 36;
      y[b * 9216 + n * 36 + sp] = a[r] + bs;   // y[b][oc][spatial] f32
    }
  }
}

// ---------------------------------------------------------------------------
// Decoder GEMM: C[256xN] = A[256xK](bf16) * Bf(pre-swizzled) + bias, act.
// Same 64x64 tile / async-A / direct-B machinery. act: 0=ReLU, 1=sigmoid.
// ---------------------------------------------------------------------------
__global__ void __launch_bounds__(256) gemm_bf16_wmma(
    const unsigned short* __restrict__ A, int lda,
    const unsigned short* __restrict__ Bf, int nt_total,
    const float* __restrict__ bias,
    float* __restrict__ outF, unsigned short* __restrict__ outH,
    int N, int ldh, int K, int act) {
  __shared__ __align__(16) unsigned short As[2][64 * 32];
  const int tid  = threadIdx.x;
  const int m0   = blockIdx.x * 64;
  const int nblk = blockIdx.y;
  const int lane = tid & 31;
  const int wave = tid >> 5;
  const int wm = wave & 3, wn = wave >> 2;
  const int ar = tid >> 2, ac = tid & 3;
  v8f acc0 = {}, acc1 = {};

  const unsigned ldsA0 = (unsigned)(size_t)&As[0][ar * 32 + ac * 8];
  const unsigned ldsA1 = (unsigned)(size_t)&As[1][ar * 32 + ac * 8];
  const unsigned short* gA = A + (m0 + ar) * lda + ac * 8;
  const int arow = wm * 16 + (lane & 15);
  const int KC = K >> 5;

  async_load_b128(ldsA0, gA);
  wait_async0();
  __syncthreads();

  for (int kc = 0; kc < KC; ++kc) {
    const unsigned short* cur = As[kc & 1];
    if (kc + 1 < KC)
      async_load_b128((kc & 1) ? ldsA0 : ldsA1, gA + (kc + 1) * 32);
    const FragBF fa = load_fragA(cur, arow, lane);
    const unsigned short* wb =
        Bf + ((size_t)kc * nt_total + nblk * 4 + wn * 2) * 512 + lane * 16;
    const FragBF fb0 = load_fragB(wb);
    const FragBF fb1 = load_fragB(wb + 512);
    acc0 = __builtin_amdgcn_wmma_f32_16x16x32_bf16(
        false, fa.v, false, fb0.v, (short)0, acc0, false, false);
    acc1 = __builtin_amdgcn_wmma_f32_16x16x32_bf16(
        false, fa.v, false, fb1.v, (short)0, acc1, false, false);
    wait_async0();
    __syncthreads();
  }

  const int nt0 = nblk * 4 + wn * 2;
#pragma unroll
  for (int s = 0; s < 2; ++s) {
    const int n = (nt0 + s) * 16 + (lane & 15);
    if (n >= N) continue;
    const float bs = bias[n];
    const v8f a = s ? acc1 : acc0;
#pragma unroll
    for (int r = 0; r < 8; ++r) {
      const int m = m0 + wm * 16 + r + ((lane < 16) ? 0 : 8);
      float x = a[r] + bs;
      x = (act == 0) ? fmaxf(x, 0.0f) : (1.0f / (1.0f + expf(-x)));
      if (outF) outF[m * N + n] = x;
      if (outH) outH[m * ldh + n] = f32_to_bf16(x);
    }
  }
}

// ---------------------------------------------------------------------------
// squash over primary capsules: u = y.reshape(B, 1152, 8); row-wise squash.
// ---------------------------------------------------------------------------
__global__ void squash_u_kernel(const float* __restrict__ y,
                                float* __restrict__ usq) {
  const int idx = blockIdx.x * 256 + threadIdx.x;   // over B*P = 294912
  const float* u = y + (size_t)idx * 8;
  float v[8], sq = 0.0f;
#pragma unroll
  for (int k = 0; k < 8; ++k) { v[k] = u[k]; sq = fmaf(v[k], v[k], sq); }
  const float scale = (sq / (1.0f + sq)) / sqrtf(sq + 1e-7f);
#pragma unroll
  for (int k = 0; k < 8; ++k) usq[(size_t)idx * 8 + k] = v[k] * scale;
}

// u_hat[b,p,j,d] = sum_k W[p,j,d,k] * usq[b,p,k]   (block = p, 160 threads)
__global__ void uhat_kernel(const float* __restrict__ W,
                            const float* __restrict__ usq,
                            float* __restrict__ uhat) {
  const int p = blockIdx.x, t = threadIdx.x;        // t = j*16 + d
  float wr[8];
#pragma unroll
  for (int k = 0; k < 8; ++k) wr[k] = W[((size_t)p * 160 + t) * 8 + k];
  for (int b = 0; b < 256; ++b) {
    const float* u = usq + ((size_t)b * 1152 + p) * 8;
    float a = 0.0f;
#pragma unroll
    for (int k = 0; k < 8; ++k) a = fmaf(wr[k], u[k], a);
    uhat[((size_t)b * 1152 + p) * 160 + t] = a;
  }
}

// ---------------------------------------------------------------------------
// Dynamic routing (5 iters) + margin loss + argmax + masked decoder input.
// ---------------------------------------------------------------------------
__global__ void __launch_bounds__(256) routing_kernel(
    const float* __restrict__ uhat, const int* __restrict__ labels,
    float* __restrict__ out_idx, unsigned short* __restrict__ dec_in,
    float* __restrict__ margin_acc) {
  __shared__ float bij[1152 * 10];
  __shared__ float sj[160];
  __shared__ float vv[160];
  __shared__ int sidx;
  const int bb = blockIdx.x, tid = threadIdx.x;
  const float* U = uhat + (size_t)bb * 1152 * 160;

  for (int i = tid; i < 11520; i += 256) bij[i] = 0.0f;

  for (int it = 0; it < 5; ++it) {
    __syncthreads();
    if (tid < 160) {                       // s_j = sum_p softmax_j(b) * u_hat
      const int j = tid >> 4;
      float acc = 0.0f;
      for (int p = 0; p < 1152; ++p) {
        const float* bp = &bij[p * 10];
        float mx = bp[0];
#pragma unroll
        for (int q = 1; q < 10; ++q) mx = fmaxf(mx, bp[q]);
        float den = 0.0f;
#pragma unroll
        for (int q = 0; q < 10; ++q) den += __expf(bp[q] - mx);
        const float c = __expf(bp[j] - mx) / den;
        acc = fmaf(c, U[p * 160 + tid], acc);
      }
      sj[tid] = acc;
    }
    __syncthreads();
    if (tid < 10) {                        // v = squash(s_j)
      float sq = 0.0f;
#pragma unroll
      for (int d = 0; d < 16; ++d) { const float x = sj[tid * 16 + d]; sq = fmaf(x, x, sq); }
      const float scale = (sq / (1.0f + sq)) / sqrtf(sq + 1e-7f);
#pragma unroll
      for (int d = 0; d < 16; ++d) vv[tid * 16 + d] = sj[tid * 16 + d] * scale;
    }
    __syncthreads();
    for (int p = tid; p < 1152; p += 256) {  // b += <v, u_hat>
      const float* up = U + p * 160;
#pragma unroll
      for (int j = 0; j < 10; ++j) {
        float a = 0.0f;
#pragma unroll
        for (int d = 0; d < 16; ++d) a = fmaf(vv[j * 16 + d], up[j * 16 + d], a);
        bij[p * 10 + j] += a;
      }
    }
  }
  __syncthreads();

  if (tid == 0) {
    int best = 0; float bestc = -1.0f, mloss = 0.0f;
    const int lab = labels[bb];
    for (int j = 0; j < 10; ++j) {
      float sq = 0.0f;
      for (int d = 0; d < 16; ++d) { const float x = vv[j * 16 + d]; sq = fmaf(x, x, sq); }
      const float cls = sqrtf(sq);
      if (cls > bestc) { bestc = cls; best = j; }
      const float vk = sqrtf(sq + 1e-7f);
      float cr = fmaxf(0.9f - vk, 0.0f); cr *= cr;
      float ic = fmaxf(vk - 0.1f, 0.0f); ic *= ic;
      mloss += (j == lab) ? cr : 0.5f * ic;
    }
    atomicAdd(margin_acc, mloss);
    out_idx[bb] = (float)best;
    sidx = best;
  }
  __syncthreads();
  if (tid < 160) {
    const int j = tid >> 4;
    dec_in[bb * 160 + tid] = f32_to_bf16(j == sidx ? vv[tid] : 0.0f);
  }
}

// Reconstruction loss partial sums (784 blocks x 256 threads == 200704 elems)
__global__ void recon_kernel(const float* __restrict__ dec_out,
                             const float* __restrict__ input,
                             float* __restrict__ recon_acc) {
  __shared__ float red[256];
  const int i = blockIdx.x * 256 + threadIdx.x;
  const float d = dec_out[i] - input[i];
  red[threadIdx.x] = d * d;
  __syncthreads();
  for (int s = 128; s > 0; s >>= 1) {
    if (threadIdx.x < s) red[threadIdx.x] += red[threadIdx.x + s];
    __syncthreads();
  }
  if (threadIdx.x == 0) atomicAdd(recon_acc, red[0]);
}

__global__ void init_scalars_kernel(float* acc) {
  if (threadIdx.x < 2) acc[threadIdx.x] = 0.0f;
}

__global__ void finalize_kernel(const float* __restrict__ acc,
                                float* __restrict__ out) {
  out[0] = acc[0] / 256.0f + 0.0005f * (acc[1] / (256.0f * 784.0f));
}

// ---------------------------------------------------------------------------
extern "C" void kernel_launch(void* const* d_in, const int* in_sizes, int n_in,
                              void* d_out, int out_size, void* d_ws, size_t ws_size,
                              hipStream_t stream) {
  (void)in_sizes; (void)n_in; (void)out_size;
  const float* input   = (const float*)d_in[0];
  const int*   labels  = (const int*)  d_in[1];
  const float* conv1_w = (const float*)d_in[2];
  const float* conv1_b = (const float*)d_in[3];
  const float* conv2_w = (const float*)d_in[4];
  const float* conv2_b = (const float*)d_in[5];
  const float* Wmat    = (const float*)d_in[6];
  const float* dw1 = (const float*)d_in[7];
  const float* db1 = (const float*)d_in[8];
  const float* dw2 = (const float*)d_in[9];
  const float* db2 = (const float*)d_in[10];
  const float* dw3 = (const float*)d_in[11];
  const float* db3 = (const float*)d_in[12];
  float* out = (float*)d_out;

  char* ws = (char*)d_ws;
  size_t off = 0;
  auto carve = [&](size_t bytes) {
    size_t o = off; off += (bytes + 255) & ~(size_t)255; return o;
  };
  unsigned short* xT   = (unsigned short*)(ws + carve((size_t)256 * 400 * 256 * 2));
  unsigned short* wp2  = (unsigned short*)(ws + carve((size_t)81 * 256 * 256 * 2));
  float* y     = (float*)(ws + carve((size_t)256 * 9216 * 4));
  float* usq   = (float*)(ws + carve((size_t)256 * 9216 * 4));
  float* uhat  = (float*)(ws + carve((size_t)256 * 1152 * 160 * 4));
  unsigned short* dec_in = (unsigned short*)(ws + carve((size_t)256 * 160 * 2));
  unsigned short* h1   = (unsigned short*)(ws + carve((size_t)256 * 512 * 2));
  unsigned short* h2   = (unsigned short*)(ws + carve((size_t)256 * 1024 * 2));
  float* dec_out = (float*)(ws + carve((size_t)256 * 784 * 4));
  unsigned short* wpd1 = (unsigned short*)(ws + carve((size_t)160 * 512 * 2));
  unsigned short* wpd2 = (unsigned short*)(ws + carve((size_t)512 * 1024 * 2));
  unsigned short* wpd3 = (unsigned short*)(ws + carve((size_t)1024 * 832 * 2));
  float* scal = (float*)(ws + carve(2 * sizeof(float)));
  if (ws_size < off) return;   // workspace too small: do nothing (deterministic)

  init_scalars_kernel<<<1, 64, 0, stream>>>(scal);
  conv1_kernel<<<256, 256, 0, stream>>>(input, conv1_w, conv1_b, xT);
  pack_w2_kernel<<<20736, 256, 0, stream>>>(conv2_w, wp2);
  pack_dec_kernel<<<(160 * 512  + 255) / 256, 256, 0, stream>>>(dw1, wpd1, 512, 160, 512);
  pack_dec_kernel<<<(512 * 1024 + 255) / 256, 256, 0, stream>>>(dw2, wpd2, 1024, 512, 1024);
  pack_dec_kernel<<<(1024 * 832 + 255) / 256, 256, 0, stream>>>(dw3, wpd3, 784, 1024, 832);

  conv2_wmma_kernel<<<dim3(144, 4), 256, 0, stream>>>(xT, wp2, conv2_b, y);
  squash_u_kernel<<<1152, 256, 0, stream>>>(y, usq);
  uhat_kernel<<<1152, 160, 0, stream>>>(Wmat, usq, uhat);
  routing_kernel<<<256, 256, 0, stream>>>(uhat, labels, out, dec_in, scal);

  gemm_bf16_wmma<<<dim3(4, 8),  256, 0, stream>>>(dec_in, 160, wpd1, 32, db1,
                                                  nullptr, h1, 512, 512, 160, 0);
  gemm_bf16_wmma<<<dim3(4, 16), 256, 0, stream>>>(h1, 512, wpd2, 64, db2,
                                                  nullptr, h2, 1024, 1024, 512, 0);
  gemm_bf16_wmma<<<dim3(4, 13), 256, 0, stream>>>(h2, 1024, wpd3, 52, db3,
                                                  dec_out, nullptr, 784, 0, 1024, 1);

  recon_kernel<<<784, 256, 0, stream>>>(dec_out, input, scal + 1);
  finalize_kernel<<<1, 1, 0, stream>>>(scal, out + 256);
}
